// HybridGraphLearner_42571715838196
// MI455X (gfx1250) — compile-verified, matrix operand order checked by reading, not run on hardware
//
#include <hip/hip_runtime.h>
#include <hip/hip_bf16.h>
#include <math.h>

#define B_ 16
#define C_ 128
#define N_ 600
#define T_ 12
#define NT_ (N_*T_)      // 7200
#define BT_ (B_*T_)      // 192
#define NN_ (N_*N_)      // 360000
#define P_  (B_*T_*N_)   // 115200

typedef __attribute__((ext_vector_type(16))) __bf16 v16bf;
typedef __attribute__((ext_vector_type(8)))  float  v8f;

__device__ __forceinline__ v8f wmma_bf16(v16bf a, v16bf b, v8f c) {
  // 8 args: (neg_a, A, neg_b, B, c_mod, C, reuse_a, reuse_b)
  return __builtin_amdgcn_wmma_f32_16x16x32_bf16(false, a, false, b, (short)0, c, false, false);
}

// Build a 16x32 (A) / 32x16 (B) bf16 fragment for wave32 WMMA.
// Lane l (l<16): index rc = l, K = {0..7, 16..23}; lanes 16..31: same rc, K+8.
template <class F>
__device__ __forceinline__ v16bf mk_frag(int lane, F f) {
  const int rc = lane & 15;
  const int kb = (lane >> 4) * 8;
  v16bf v;
#pragma unroll
  for (int e = 0; e < 8; ++e) {
    v[e]     = (__bf16)f(rc, kb + e);
    v[e + 8] = (__bf16)f(rc, kb + 16 + e);
  }
  return v;
}

// 32x32 output tile per wave: 2x2 register blocking -> 4 WMMAs per 4 fragment loads.
// fa(r,k): A element at tile-row r (0..31), absolute k
// fb(k,c): B element at absolute k, tile-col c (0..31)
// fs(r,c,v): store epilogue at tile-relative (r,c)
template <class FA, class FB, class FS>
__device__ __forceinline__ void gemm22(int lane, int K, FA fa, FB fb, FS fs) {
  v8f c00 = {}, c01 = {}, c10 = {}, c11 = {};
  for (int k0 = 0; k0 < K; k0 += 32) {
    v16bf a0 = mk_frag(lane, [&](int r, int k) { return fa(r,      k0 + k); });
    v16bf a1 = mk_frag(lane, [&](int r, int k) { return fa(16 + r, k0 + k); });
    v16bf b0 = mk_frag(lane, [&](int c, int k) { return fb(k0 + k, c); });
    v16bf b1 = mk_frag(lane, [&](int c, int k) { return fb(k0 + k, 16 + c); });
    c00 = wmma_bf16(a0, b0, c00);
    c01 = wmma_bf16(a0, b1, c01);
    c10 = wmma_bf16(a1, b0, c10);
    c11 = wmma_bf16(a1, b1, c11);
  }
  const int col = lane & 15, hi = lane >> 4;
#pragma unroll
  for (int v = 0; v < 8; ++v) {
    int r = v + 8 * hi;
    fs(r,      col,      c00[v]);
    fs(r,      col + 16, c01[v]);
    fs(r + 16, col,      c10[v]);
    fs(r + 16, col + 16, c11[v]);
  }
}

// ---------------- 1x1 conv over [B,C,N,T]: out[o,p] = sum_c W[o,c]*in[c,p] + bias ----
__global__ void k_conv_bcnt(const float* __restrict__ in, const float* __restrict__ W,
                            const float* __restrict__ bias, float* __restrict__ out) {
  const int lane = threadIdx.x;
  const int p0 = blockIdx.x * 32;
  const int o0 = blockIdx.y * 32;
  const int b  = blockIdx.z;
  const float* inb = in + (size_t)b * C_ * NT_;
  float* outb = out + (size_t)b * C_ * NT_;
  __builtin_prefetch(W + o0 * C_, 0, 0);             // weights: tiny, maximally reused
  gemm22(lane, C_,
    [&](int r, int k)          { return W[(o0 + r) * C_ + k]; },
    [&](int k, int c)          { return inb[k * NT_ + p0 + c]; },
    [&](int r, int c, float v) { outb[(o0 + r) * NT_ + p0 + c] = v + bias[o0 + r]; });
}

// ---------------- generic row GEMM: out[p,f] = sum_c in[p,c] * W[c,f] ----------------
__global__ void k_rowgemm(const float* __restrict__ in, const float* __restrict__ W,
                          float* __restrict__ out) {
  const int lane = threadIdx.x;
  const int p0 = blockIdx.x * 32;
  const int f0 = blockIdx.y * 32;
  __builtin_prefetch(W, 0, 0);
  gemm22(lane, C_,
    [&](int r, int k)          { return in[(size_t)(p0 + r) * C_ + k]; },
    [&](int k, int c)          { return W[k * C_ + f0 + c]; },
    [&](int r, int c, float v) { out[(size_t)(p0 + r) * C_ + f0 + c] = v; });
}

// ---------------- Wh = transpose(xc)->[B,T,N,C] @ W : fused transpose-GEMM ----------
__global__ void k_whx(const float* __restrict__ xc, const float* __restrict__ W,
                      float* __restrict__ wh) {
  const int lane = threadIdx.x;
  const int n0 = blockIdx.x * 32;
  const int f0 = blockIdx.y * 32;
  const int bt = blockIdx.z;
  const int b = bt / T_, t = bt % T_;
  const float* xb = xc + (size_t)b * C_ * NT_;
  gemm22(lane, C_,
    [&](int r, int k) {
      int n = n0 + r;
      return (n < N_) ? xb[(k * N_ + n) * T_ + t] : 0.f;
    },
    [&](int k, int c)          { return W[k * C_ + f0 + c]; },
    [&](int r, int c, float v) {
      int n = n0 + r;
      if (n < N_) wh[((size_t)bt * N_ + n) * C_ + f0 + c] = v;
    });
}

// ---------------- time-sum: xr[b,c,n] = sum_t xc[b,c,n,t] ---------------------------
__global__ void k_tsum(const float* __restrict__ xc, float* __restrict__ xr) {
  int i = blockIdx.x * blockDim.x + threadIdx.x;
  if (i >= B_ * C_ * N_) return;
  const float* p = xc + (size_t)i * T_;
  float s = 0.f;
#pragma unroll
  for (int t = 0; t < T_; ++t) s += p[t];
  xr[i] = s;
}

// ---------------- E_d[b,n,m] = tanh(sum_c xr[b,c,n]*E_s[c,m]) ------------------------
__global__ void k_ed(const float* __restrict__ xr, const float* __restrict__ Es,
                     float* __restrict__ Ed) {
  const int lane = threadIdx.x;
  const int n0 = blockIdx.x * 32;
  const int m0 = blockIdx.y * 32;
  const int b = blockIdx.z;
  const float* xb = xr + (size_t)b * C_ * N_;
  gemm22(lane, C_,
    [&](int r, int k) {
      int n = n0 + r;
      return (n < N_) ? xb[k * N_ + n] : 0.f;
    },
    [&](int k, int c) {
      int m = m0 + c;
      return (m < N_) ? Es[k * N_ + m] : 0.f;
    },
    [&](int r, int c, float v) {
      int n = n0 + r, m = m0 + c;
      if (n < N_ && m < N_) Ed[((size_t)b * N_ + n) * N_ + m] = tanhf(v);
    });
}

// ---------------- S = relu(Ed @ Ed^T / sqrt(C)) --------------------------------------
__global__ void k_gram(const float* __restrict__ Ed, float* __restrict__ S) {
  const int lane = threadIdx.x;
  const int n0 = blockIdx.x * 32;
  const int m0 = blockIdx.y * 32;
  const int b = blockIdx.z;
  const float* Eb = Ed + (size_t)b * NN_;
  gemm22(lane, N_,
    [&](int r, int k) {
      int n = n0 + r;
      return (n < N_ && k < N_) ? Eb[n * N_ + k] : 0.f;
    },
    [&](int k, int c) {
      int m = m0 + c;
      return (m < N_ && k < N_) ? Eb[m * N_ + k] : 0.f;
    },
    [&](int r, int c, float v) {
      int n = n0 + r, m = m0 + c;
      if (n < N_ && m < N_) {
        float s = v * 0.088388347648318447f;  // 1/sqrt(128)
        S[((size_t)b * N_ + n) * N_ + m] = s > 0.f ? s : 0.f;
      }
    });
}

// ---------------- row softmax of S + mean over batch (atomic accumulate) -------------
__global__ void k_softmax_mean(const float* __restrict__ S, float* __restrict__ Asum) {
  const int row = blockIdx.x;       // b*N + n
  const int n = row % N_;
  const int tid = threadIdx.x;
  const float* r = S + (size_t)row * N_;
  __shared__ float red[128];
  float mx = -1e30f;
  for (int m = tid; m < N_; m += 128) mx = fmaxf(mx, r[m]);
  red[tid] = mx; __syncthreads();
  for (int s = 64; s > 0; s >>= 1) { if (tid < s) red[tid] = fmaxf(red[tid], red[tid + s]); __syncthreads(); }
  mx = red[0]; __syncthreads();
  float sum = 0.f;
  for (int m = tid; m < N_; m += 128) sum += __expf(r[m] - mx);
  red[tid] = sum; __syncthreads();
  for (int s = 64; s > 0; s >>= 1) { if (tid < s) red[tid] += red[tid + s]; __syncthreads(); }
  sum = red[0];
  float inv = 1.f / (sum * (float)B_);
  for (int m = tid; m < N_; m += 128)
    atomicAdd(&Asum[n * N_ + m], __expf(r[m] - mx) * inv);
}

// ---------------- A_adp: per-row softmax(relu(nv1@nv2)) ------------------------------
__global__ void k_adp(const float* __restrict__ nv1, const float* __restrict__ nv2,
                      float* __restrict__ Aadp) {
  const int i = blockIdx.x;
  const int tid = threadIdx.x;
  __shared__ float e[N_];
  __shared__ float red[128];
  for (int j = tid; j < N_; j += 128) {
    float s = 0.f;
#pragma unroll
    for (int q = 0; q < 6; ++q) s += nv1[i * 6 + q] * nv2[q * N_ + j];
    e[j] = s > 0.f ? s : 0.f;
  }
  __syncthreads();
  float mx = -1e30f;
  for (int j = tid; j < N_; j += 128) mx = fmaxf(mx, e[j]);
  red[tid] = mx; __syncthreads();
  for (int s = 64; s > 0; s >>= 1) { if (tid < s) red[tid] = fmaxf(red[tid], red[tid + s]); __syncthreads(); }
  mx = red[0]; __syncthreads();
  float sum = 0.f;
  for (int j = tid; j < N_; j += 128) sum += __expf(e[j] - mx);
  red[tid] = sum; __syncthreads();
  for (int s = 64; s > 0; s >>= 1) { if (tid < s) red[tid] += red[tid + s]; __syncthreads(); }
  sum = red[0];
  for (int j = tid; j < N_; j += 128) Aadp[i * N_ + j] = __expf(e[j] - mx) / sum;
}

// ---------------- x_gcn spatial: out[b,c,m,t] = sum_n xc[b,c,n,t]*Aadp[n,m] ---------
__global__ void k_spatial(const float* __restrict__ xc, const float* __restrict__ Aadp,
                          float* __restrict__ out) {
  const int lane = threadIdx.x;
  const int m0 = blockIdx.x * 32;
  const int c0 = blockIdx.y * 32;
  const int bt = blockIdx.z;
  const int b = bt / T_, t = bt % T_;
  const float* xb = xc + (size_t)b * C_ * NT_;
  float* ob = out + (size_t)b * C_ * NT_;
  gemm22(lane, N_,
    [&](int r, int k) {
      return (k < N_) ? xb[((c0 + r) * N_ + k) * T_ + t] : 0.f;
    },
    [&](int k, int c) {
      int m = m0 + c;
      return (k < N_ && m < N_) ? Aadp[k * N_ + m] : 0.f;
    },
    [&](int r, int c, float v) {
      int m = m0 + c;
      if (m < N_) ob[((c0 + r) * N_ + m) * T_ + t] = v;
    });
}

// ---------------- attention rank-1 projections s1/s2 ---------------------------------
__global__ void k_sv(const float* __restrict__ wh, const float* __restrict__ avec,
                     float* __restrict__ s1, float* __restrict__ s2) {
  const int wave = threadIdx.x >> 5;
  const int lane = threadIdx.x & 31;
  const int p = blockIdx.x * 4 + wave;
  const float* r = wh + (size_t)p * C_;
  float v1 = 0.f, v2 = 0.f;
#pragma unroll
  for (int q = 0; q < 4; ++q) {
    int c = lane + 32 * q;
    float h = r[c];
    v1 += h * avec[c];
    v2 += h * avec[C_ + c];
  }
  for (int off = 16; off > 0; off >>= 1) {
    v1 += __shfl_down(v1, off, 32);
    v2 += __shfl_down(v2, off, 32);
  }
  if (lane == 0) { s1[p] = v1; s2[p] = v2; }
}

// ---------------- per-row softmax stats (max, denom) without materializing e ---------
__global__ void k_stats(const float* __restrict__ s1, const float* __restrict__ s2,
                        const float* __restrict__ Adyn,
                        float* __restrict__ mx, float* __restrict__ dn) {
  const int p = blockIdx.x;
  const int n = p % N_;
  const int btb = (p / N_) * N_;
  const int tid = threadIdx.x;
  __shared__ float red[128];
  const float s1v = s1[p];
  float m = -3.0e38f;
  for (int j = tid; j < N_; j += 128) {
    float e = s1v + s2[btb + j];
    e = e > 0.f ? e : 0.2f * e;
    float v = (Adyn[n * N_ + j] > 0.f) ? e : -9.0e15f;
    m = fmaxf(m, v);
  }
  red[tid] = m; __syncthreads();
  for (int s = 64; s > 0; s >>= 1) { if (tid < s) red[tid] = fmaxf(red[tid], red[tid + s]); __syncthreads(); }
  m = red[0]; __syncthreads();
  float sum = 0.f;
  for (int j = tid; j < N_; j += 128) {
    float e = s1v + s2[btb + j];
    e = e > 0.f ? e : 0.2f * e;
    float v = (Adyn[n * N_ + j] > 0.f) ? e : -9.0e15f;
    sum += __expf(v - m);
  }
  red[tid] = sum; __syncthreads();
  for (int s = 64; s > 0; s >>= 1) { if (tid < s) red[tid] += red[tid + s]; __syncthreads(); }
  if (tid == 0) { mx[p] = m; dn[p] = red[0]; }
}

// ---------------- fused attention aggregate: elu(softmax(e) @ Wh) --------------------
__global__ void k_agg(const float* __restrict__ wh, const float* __restrict__ s1,
                      const float* __restrict__ s2, const float* __restrict__ mx,
                      const float* __restrict__ dn, const float* __restrict__ Adyn,
                      float* __restrict__ out) {
  const int lane = threadIdx.x;
  const int n0 = blockIdx.x * 32;
  const int f0 = blockIdx.y * 32;
  const int bt = blockIdx.z;
  const size_t pb = (size_t)bt * N_;
  gemm22(lane, N_,
    [&](int r, int k) {
      int n = n0 + r;
      if (n >= N_ || k >= N_) return 0.f;
      float e = s1[pb + n] + s2[pb + k];
      e = e > 0.f ? e : 0.2f * e;
      float v = (Adyn[n * N_ + k] > 0.f) ? e : -9.0e15f;
      return __expf(v - mx[pb + n]);      // unnormalized probability in [0,1]
    },
    [&](int k, int c) {
      return (k < N_) ? wh[(pb + k) * C_ + f0 + c] : 0.f;
    },
    [&](int r, int c, float v) {
      int n = n0 + r;
      if (n < N_) {
        float val = v / dn[pb + n];
        val = val > 0.f ? val : __expf(val) - 1.f;   // elu
        out[(pb + n) * C_ + f0 + c] = val;
      }
    });
}

// ---------------- conv with fused [B,T,N,C] -> [B,C,N,T] transpose -------------------
__global__ void k_conv_btnc(const float* __restrict__ g, const float* __restrict__ W,
                            const float* __restrict__ bias, float* __restrict__ out) {
  const int lane = threadIdx.x;
  const int q0 = blockIdx.x * 32;     // q = n*T + t
  const int o0 = blockIdx.y * 32;
  const int b = blockIdx.z;
  float* ob = out + (size_t)b * C_ * NT_;
  __builtin_prefetch(W + o0 * C_, 0, 0);
  gemm22(lane, C_,
    [&](int r, int k) { return W[(o0 + r) * C_ + k]; },
    [&](int k, int c) {
      int q = q0 + c;
      int n = q / T_, t = q - n * T_;
      return g[(((size_t)b * T_ + t) * N_ + n) * C_ + k];
    },
    [&](int r, int c, float v) { ob[(o0 + r) * NT_ + q0 + c] = v + bias[o0 + r]; });
}

// ---------------- elementwise helpers ------------------------------------------------
__global__ void k_zero(float* __restrict__ p, int n) {
  int i = blockIdx.x * blockDim.x + threadIdx.x;
  if (i < n) p[i] = 0.f;
}
__global__ void k_thresh(const float* __restrict__ a, float* __restrict__ d, int n) {
  int i = blockIdx.x * blockDim.x + threadIdx.x;
  if (i < n) d[i] = a[i] > 0.5f ? 1.f : 0.f;
}
__global__ void k_combine(const float* __restrict__ gcn, const float* __restrict__ gat,
                          const float* __restrict__ alpha, float* __restrict__ out, int n) {
  int i = blockIdx.x * blockDim.x + threadIdx.x;
  if (i >= n) return;
  float s = 1.f / (1.f + __expf(-alpha[0]));
  out[i] = s * gcn[i] + (1.f - s) * gat[i];
}
__global__ void k_final(const float* __restrict__ f, const float* __restrict__ emb,
                        const float* __restrict__ x, float* __restrict__ out, int n) {
  int i = blockIdx.x * blockDim.x + threadIdx.x;
  if (i >= n) return;
  int rem = i % (C_ * NT_);
  out[i] = f[i] * emb[rem] + x[i];
}

extern "C" void kernel_launch(void* const* d_in, const int* in_sizes, int n_in,
                              void* d_out, int out_size, void* d_ws, size_t ws_size,
                              hipStream_t stream) {
  (void)in_sizes; (void)n_in; (void)out_size; (void)ws_size;
  const float* x      = (const float*)d_in[0];
  const float* W_conv = (const float*)d_in[1];
  const float* b_conv = (const float*)d_in[2];
  const float* E_s    = (const float*)d_in[3];
  const float* nv1    = (const float*)d_in[4];
  const float* nv2    = (const float*)d_in[5];
  const float* W_gcn  = (const float*)d_in[6];
  const float* b_gcn  = (const float*)d_in[7];
  const float* W_cgcn = (const float*)d_in[8];
  const float* b_cgcn = (const float*)d_in[9];
  const float* gat_W1 = (const float*)d_in[10];
  const float* gat_a1 = (const float*)d_in[11];
  const float* gat_Wo = (const float*)d_in[12];
  const float* gat_ao = (const float*)d_in[13];
  const float* W_cgat = (const float*)d_in[14];
  const float* b_cgat = (const float*)d_in[15];
  const float* alpha  = (const float*)d_in[16];
  const float* W_fus  = (const float*)d_in[17];
  const float* b_fus  = (const float*)d_in[18];
  const float* emb    = (const float*)d_in[19];
  float* out = (float*)d_out;

  float* ws = (float*)d_ws;
  const size_t SZX = (size_t)B_ * C_ * NT_;       // 14,745,600
  float* xc   = ws;                               // conv1 output [B,C,N,T]
  float* tmpA = xc + SZX;                         // gcn spatial / gcn final
  float* xg   = tmpA + SZX;                       // conv ping / gat conv out
  float* wh   = xg + SZX;                         // Wh (reused layer2, combine)
  float* h1   = wh + SZX;                         // gat hidden / fus conv out
  float* Sb   = h1 + SZX;                         // S [B,N,N]
  float* Ed   = Sb + (size_t)B_ * NN_;            // E_d [B,N,N]
  float* Aadp = Ed + (size_t)B_ * NN_;
  float* Asum = Aadp + NN_;
  float* Adyn = Asum + NN_;
  float* xr   = Adyn + NN_;                       // [B,C,N]
  float* s1   = xr + (size_t)B_ * C_ * N_;
  float* s2   = s1 + P_;
  float* mxb  = s2 + P_;
  float* dnb  = mxb + P_;

  dim3 w32(32);
  const int N32 = (N_ + 31) / 32;                 // 19

  k_zero<<<(NN_ + 255) / 256, 256, 0, stream>>>(Asum, NN_);
  k_adp<<<N_, 128, 0, stream>>>(nv1, nv2, Aadp);

  k_conv_bcnt<<<dim3(NT_ / 32, C_ / 32, B_), w32, 0, stream>>>(x, W_conv, b_conv, xc);
  k_tsum<<<(B_ * C_ * N_ + 255) / 256, 256, 0, stream>>>(xc, xr);
  k_ed<<<dim3(N32, N32, B_), w32, 0, stream>>>(xr, E_s, Ed);
  k_gram<<<dim3(N32, N32, B_), w32, 0, stream>>>(Ed, Sb);
  k_softmax_mean<<<B_ * N_, 128, 0, stream>>>(Sb, Asum);
  k_thresh<<<(NN_ + 255) / 256, 256, 0, stream>>>(Asum, Adyn, NN_);

  // GCN branch
  k_spatial<<<dim3(N32, C_ / 32, BT_), w32, 0, stream>>>(xc, Aadp, tmpA);
  k_conv_bcnt<<<dim3(NT_ / 32, C_ / 32, B_), w32, 0, stream>>>(tmpA, W_gcn, b_gcn, xg);
  k_conv_bcnt<<<dim3(NT_ / 32, C_ / 32, B_), w32, 0, stream>>>(xg, W_cgcn, b_cgcn, tmpA);

  // GAT layer 1
  k_whx<<<dim3(N32, C_ / 32, BT_), w32, 0, stream>>>(xc, gat_W1, wh);
  k_sv<<<P_ / 4, 128, 0, stream>>>(wh, gat_a1, s1, s2);
  k_stats<<<P_, 128, 0, stream>>>(s1, s2, Adyn, mxb, dnb);
  k_agg<<<dim3(N32, C_ / 32, BT_), w32, 0, stream>>>(wh, s1, s2, mxb, dnb, Adyn, h1);

  // GAT layer 2 (xc free -> holds layer-2 output)
  k_rowgemm<<<dim3(P_ / 32, C_ / 32), w32, 0, stream>>>(h1, gat_Wo, wh);
  k_sv<<<P_ / 4, 128, 0, stream>>>(wh, gat_ao, s1, s2);
  k_stats<<<P_, 128, 0, stream>>>(s1, s2, Adyn, mxb, dnb);
  k_agg<<<dim3(N32, C_ / 32, BT_), w32, 0, stream>>>(wh, s1, s2, mxb, dnb, Adyn, xc);

  // transpose-back + conv, fuse, final
  k_conv_btnc<<<dim3(NT_ / 32, C_ / 32, B_), w32, 0, stream>>>(xc, W_cgat, b_cgat, xg);
  k_combine<<<(int)((SZX + 255) / 256), 256, 0, stream>>>(tmpA, xg, alpha, wh, (int)SZX);
  k_conv_bcnt<<<dim3(NT_ / 32, C_ / 32, B_), w32, 0, stream>>>(wh, W_fus, b_fus, h1);
  k_final<<<(int)((SZX + 255) / 256), 256, 0, stream>>>(h1, emb, x, out, (int)SZX);
}
